// NoskipViTEncoder_81853486727748
// MI455X (gfx1250) — compile-verified, matrix operand order checked by reading, not run on hardware
//
#include <hip/hip_runtime.h>
#include <hip/hip_bf16.h>
#include <math.h>

// ---------------- problem constants ----------------
#define BB      2
#define CC      64
#define SS      112
#define KWIN    7
#define NHEADS  4
#define DHD     16
#define HWPIX   (SS * SS)        // 12544
#define NTOK    (BB * HWPIX)     // 25088
#define TOKC    (NTOK * CC)      // 1,605,632 floats per token-major tensor

// half-precision weight pool offsets (in _Float16 elements)
#define OFF_WS16   0                      // conv_w_s  [co][tap][ci]  64*576
#define OFF_WL16   36864                  // conv_w_l
#define OFF_WQ16   73728                  // Wq [k][n] 64x64
#define OFF_WK16   77824
#define OFF_WV16   81920
#define OFF_WO16   86016
#define OFF_W116   90112                  // W1 [k][n] 64x128
#define OFF_W216   98304                  // W2 [k][n] 128x64
#define NHALF      106496

typedef __attribute__((ext_vector_type(16))) _Float16 v16h;
typedef __attribute__((ext_vector_type(8)))  float    v8f;

// ---------------- WMMA fragment helpers (ISA 7.12.2 16-bit layouts) ----------
// For 16-bit A (16x32) and B (32x16) fragments, lane L holds 16 halves whose K
// indices form two contiguous runs of 8:  K = 8g..8g+7 and 16+8g..23+8g with
// g = L>>4.  Caller supplies a pointer such that element K=kk lives at
// base[kk] (contiguous-K) or base[kk*stride] (strided-K).
__device__ __forceinline__ v16h frag16(const _Float16* base) {
  const _Float16* p = base + 8 * ((threadIdx.x >> 4) & 1);
  v16h f;
#pragma unroll
  for (int t = 0; t < 8; ++t) f[t] = p[t];
#pragma unroll
  for (int t = 0; t < 8; ++t) f[8 + t] = p[16 + t];
  return f;
}

__device__ __forceinline__ v16h frag16s(const _Float16* base, int stride) {
  const _Float16* p = base + 8 * ((threadIdx.x >> 4) & 1) * stride;
  v16h f;
#pragma unroll
  for (int t = 0; t < 8; ++t) f[t] = p[t * stride];
#pragma unroll
  for (int t = 0; t < 8; ++t) f[8 + t] = p[(16 + t) * stride];
  return f;
}

__device__ __forceinline__ v8f wmma16(v16h a, v16h b, v8f c) {
  // D = A(16x32,f16) x B(32x16,f16) + C(16x16,f32)
  return __builtin_amdgcn_wmma_f32_16x16x32_f16(false, a, false, b,
                                                (short)0, c, false, false);
}

// ---------------- kernel 0: weight conversion / reordering ------------------
__global__ __launch_bounds__(256)
void prep_weights(const float* __restrict__ cws, const float* __restrict__ cwl,
                  const float* __restrict__ Wq, const float* __restrict__ Wk,
                  const float* __restrict__ Wv, const float* __restrict__ Wo,
                  const float* __restrict__ W1, const float* __restrict__ W2,
                  _Float16* __restrict__ hb) {
  int idx = blockIdx.x * 256 + threadIdx.x;
  if (idx >= NHALF) return;
  float v;
  if (idx < OFF_WQ16) {                       // conv weights: [co][ci][kh][kw] -> [co][tap][ci]
    int which = idx / 36864;
    int r = idx % 36864;
    int co = r / 576, rr = r % 576;
    int tap = rr / 64, ci = rr % 64;
    int kh = tap / 3, kw = tap % 3;
    const float* src = which ? cwl : cws;
    v = src[((co * 64 + ci) * 3 + kh) * 3 + kw];
  } else {                                    // dense mats: straight copy, [k][n] row-major
    int r = idx - OFF_WQ16;
    const float* src; int off;
    if      (r <  4096) { src = Wq; off = r; }
    else if (r <  8192) { src = Wk; off = r - 4096; }
    else if (r < 12288) { src = Wv; off = r - 8192; }
    else if (r < 16384) { src = Wo; off = r - 12288; }
    else if (r < 24576) { src = W1; off = r - 16384; }
    else                { src = W2; off = r - 24576; }
    v = src[off];
  }
  hb[idx] = (_Float16)v;
}

// ---------------- kernel 1: 3x3 conv as implicit GEMM (WMMA) ----------------
// grid = B*112*7 blocks; block = 128 threads (4 waves). Each block computes a
// 16-pixel x 64-out-channel tile. K = 9 taps * 64 ci = 18 chunks of 32.
__global__ __launch_bounds__(128)
void conv3x3_wmma(const float* __restrict__ x, const _Float16* __restrict__ wg,
                  const float* __restrict__ bias, float* __restrict__ emb) {
  __shared__ _Float16 patch[3][18][64];   // [row][wcol][ci], f16
  int blk = blockIdx.x;
  int wt = blk % 7, h = (blk / 7) % SS, b = blk / (7 * SS);
  int w0 = wt * 16;
  int tid = threadIdx.x;

  // cooperative padded-patch load, coalesced along w
  for (int i = tid; i < 3 * 18 * 64; i += 128) {
    int wc = i % 18;
    int ci = (i / 18) % 64;
    int r  = i / (18 * 64);
    int gh = h + r - 1, gw = w0 + wc - 1;
    float v = 0.f;
    if (gh >= 0 && gh < SS && gw >= 0 && gw < SS)
      v = x[(((size_t)b * 64 + ci) * SS + gh) * SS + gw];
    patch[r][wc][ci] = (_Float16)v;
  }
  __syncthreads();

  int wave = tid >> 5, lane = tid & 31;
  int co0 = wave * 16;
  int mrow = lane & 15, n = lane & 15;
  v8f acc = {};
#pragma unroll
  for (int kh = 0; kh < 3; ++kh)
#pragma unroll
    for (int kw = 0; kw < 3; ++kw)
#pragma unroll
      for (int c0 = 0; c0 < 64; c0 += 32) {
        v16h a  = frag16(&patch[kh][mrow + kw][c0]);
        v16h bf = frag16(wg + (size_t)(co0 + n) * 576 + (kh * 3 + kw) * 64 + c0);
        acc = wmma16(a, bf, acc);
      }

  float bv = bias[co0 + n];
#pragma unroll
  for (int r = 0; r < 8; ++r) {
    int m = r + 8 * ((lane >> 4) & 1);
    emb[(((size_t)b * SS + h) * SS + (w0 + m)) * 64 + co0 + n] = acc[r] + bv;
  }
}

// ---------------- kernel 2: fused LN + QKV projection (WMMA) ----------------
// grid = NTOK/16; block = 128. Wave w == head w (co0 = 16w -> one head's 16 ch).
__global__ __launch_bounds__(128)
void ln_qkv_wmma(const float* __restrict__ embq, const float* __restrict__ embkv,
                 const float* __restrict__ g_s, const float* __restrict__ b_s,
                 const float* __restrict__ g_l, const float* __restrict__ b_l,
                 const _Float16* __restrict__ wq, const _Float16* __restrict__ wk,
                 const _Float16* __restrict__ wv,
                 const float* __restrict__ bq, const float* __restrict__ bk,
                 const float* __restrict__ bv,
                 float* __restrict__ qo, float* __restrict__ ko,
                 float* __restrict__ vo) {
  __shared__ float    sq[16][64], skv[16][64];
  __shared__ _Float16 aq[16][64], akv[16][64];
  int t0 = blockIdx.x * 16;
  int tid = threadIdx.x;

  for (int i = tid; i < 16 * 64; i += 128) {
    int m = i >> 6, c = i & 63;
    sq[m][c]  = embq [(size_t)(t0 + m) * 64 + c];
    skv[m][c] = embkv[(size_t)(t0 + m) * 64 + c];
  }
  __syncthreads();

  if (tid < 32) {                 // LayerNorm, one thread per (token, q/kv)
    int m = tid & 15, which = tid >> 4;
    float* row = which ? skv[m] : sq[m];
    float s = 0.f;
    for (int c = 0; c < 64; ++c) s += row[c];
    float mean = s * (1.f / 64.f), v2 = 0.f;
    for (int c = 0; c < 64; ++c) { float d = row[c] - mean; v2 += d * d; }
    float rinv = rsqrtf(v2 * (1.f / 64.f) + 1e-5f);
    const float* gg = which ? g_l : g_s;
    const float* bb = which ? b_l : b_s;
    _Float16* dst = which ? akv[m] : aq[m];
    for (int c = 0; c < 64; ++c)
      dst[c] = (_Float16)((row[c] - mean) * rinv * gg[c] + bb[c]);
  }
  __syncthreads();

  int wave = tid >> 5, lane = tid & 31;
  int head = wave, n = lane & 15, mrow = lane & 15;
  v8f accq = {}, acck = {}, accv = {};
#pragma unroll
  for (int c0 = 0; c0 < 64; c0 += 32) {
    v16h a1 = frag16(&aq[mrow][c0]);
    v16h a2 = frag16(&akv[mrow][c0]);
    accq = wmma16(a1, frag16s(wq + c0 * 64 + head * 16 + n, 64), accq);
    acck = wmma16(a2, frag16s(wk + c0 * 64 + head * 16 + n, 64), acck);
    accv = wmma16(a2, frag16s(wv + c0 * 64 + head * 16 + n, 64), accv);
  }
#pragma unroll
  for (int r = 0; r < 8; ++r) {
    int m = r + 8 * ((lane >> 4) & 1);
    int t = t0 + m;
    int b = t / HWPIX, pix = t - b * HWPIX;
    size_t off = ((size_t)(b * NHEADS + head) * HWPIX + pix) * 16 + n;
    qo[off] = (accq[r] + bq[head * 16 + n]) * 0.25f;  // fold DH^-0.5
    ko[off] = acck[r] + bk[head * 16 + n];
    vo[off] = accv[r] + bv[head * 16 + n];
  }
}

// ---------------- kernel 3: neighborhood attention, online softmax ----------
__global__ __launch_bounds__(128)
void na_attn(const float* __restrict__ q, const float* __restrict__ k,
             const float* __restrict__ v, const float* __restrict__ rpb,
             float* __restrict__ ao) {
  int gid = blockIdx.x * 128 + threadIdx.x;
  if (gid >= BB * NHEADS * HWPIX) return;
  int b = gid / (NHEADS * HWPIX);
  int rem = gid - b * NHEADS * HWPIX;
  int head = rem / HWPIX;
  int pix = rem - head * HWPIX;
  int i = pix / SS, j = pix % SS;

  size_t hbase = (size_t)(b * NHEADS + head) * HWPIX;
  float qv[16];
  const float* qp = q + (hbase + pix) * 16;
#pragma unroll
  for (int d = 0; d < 16; ++d) qv[d] = qp[d];

  int si = i - 3; si = si < 0 ? 0 : (si > SS - KWIN ? SS - KWIN : si);
  int sj = j - 3; sj = sj < 0 ? 0 : (sj > SS - KWIN ? SS - KWIN : sj);
  const float* rb = rpb + head * (2 * KWIN - 1) * (2 * KWIN - 1);

  float mx = -1e30f, s = 0.f, acc[16];
#pragma unroll
  for (int d = 0; d < 16; ++d) acc[d] = 0.f;

  for (int di = 0; di < KWIN; ++di) {
    int ni = si + di;
    int ri = ni - i + (KWIN - 1);
    for (int dj = 0; dj < KWIN; ++dj) {
      int nj = sj + dj;
      size_t noff = (hbase + (size_t)ni * SS + nj) * 16;
      const float* kp = k + noff;
      float dot = 0.f;
#pragma unroll
      for (int d = 0; d < 16; ++d) dot += qv[d] * kp[d];  // q pre-scaled
      float l = dot + rb[ri * (2 * KWIN - 1) + (nj - j + (KWIN - 1))];
      float nm = fmaxf(mx, l);
      float sc = __expf(mx - nm);
      float p  = __expf(l - nm);
      s = s * sc + p;
      const float* vp = v + noff;
#pragma unroll
      for (int d = 0; d < 16; ++d) acc[d] = acc[d] * sc + p * vp[d];
      mx = nm;
    }
  }
  float inv = 1.f / s;
  float* op = ao + (size_t)(b * HWPIX + pix) * 64 + head * 16;
#pragma unroll
  for (int d = 0; d < 16; ++d) op[d] = acc[d] * inv;
}

// ---------------- kernel 4: output projection + residual (WMMA) -------------
__global__ __launch_bounds__(128)
void oproj_wmma(const float* __restrict__ ain, const _Float16* __restrict__ wo,
                const float* __restrict__ bo, const float* __restrict__ ekv,
                float* __restrict__ att) {
  __shared__ _Float16 a16[16][64];
  int t0 = blockIdx.x * 16, tid = threadIdx.x;
  for (int i = tid; i < 1024; i += 128) {
    int m = i >> 6, c = i & 63;
    a16[m][c] = (_Float16)ain[(size_t)(t0 + m) * 64 + c];
  }
  __syncthreads();
  int wave = tid >> 5, lane = tid & 31;
  int co0 = wave * 16, n = lane & 15, mrow = lane & 15;
  v8f acc = {};
#pragma unroll
  for (int c0 = 0; c0 < 64; c0 += 32)
    acc = wmma16(frag16(&a16[mrow][c0]), frag16s(wo + c0 * 64 + co0 + n, 64), acc);
  float bv = bo[co0 + n];
#pragma unroll
  for (int r = 0; r < 8; ++r) {
    int m = r + 8 * ((lane >> 4) & 1);
    size_t t = t0 + m;
    att[t * 64 + co0 + n] = acc[r] + bv + ekv[t * 64 + co0 + n];
  }
}

// ---------------- kernel 5: fused LN + MLP (GELU) + residual, NCHW store ----
__global__ __launch_bounds__(128)
void mlp_wmma(const float* __restrict__ att, const float* __restrict__ g_n,
              const float* __restrict__ b_n, const _Float16* __restrict__ w1,
              const float* __restrict__ b1, const _Float16* __restrict__ w2,
              const float* __restrict__ b2, float* __restrict__ outp) {
  __shared__ float    sa[16][64];
  __shared__ _Float16 xln[16][64];
  __shared__ _Float16 h1[16][128];
  int t0 = blockIdx.x * 16, tid = threadIdx.x;

  for (int i = tid; i < 1024; i += 128) {
    int m = i >> 6, c = i & 63;
    sa[m][c] = att[(size_t)(t0 + m) * 64 + c];
  }
  __syncthreads();
  if (tid < 16) {
    float* row = sa[tid];
    float s = 0.f;
    for (int c = 0; c < 64; ++c) s += row[c];
    float mean = s * (1.f / 64.f), v2 = 0.f;
    for (int c = 0; c < 64; ++c) { float d = row[c] - mean; v2 += d * d; }
    float rinv = rsqrtf(v2 * (1.f / 64.f) + 1e-5f);
    for (int c = 0; c < 64; ++c)
      xln[tid][c] = (_Float16)((row[c] - mean) * rinv * g_n[c] + b_n[c]);
  }
  __syncthreads();

  int wave = tid >> 5, lane = tid & 31;
  int n = lane & 15, mrow = lane & 15;

  // GEMM1: (16x64) x (64x128) -> GELU -> h1 (f16 in LDS)
#pragma unroll
  for (int half = 0; half < 2; ++half) {
    int co0 = wave * 32 + half * 16;
    v8f acc = {};
#pragma unroll
    for (int c0 = 0; c0 < 64; c0 += 32)
      acc = wmma16(frag16(&xln[mrow][c0]), frag16s(w1 + c0 * 128 + co0 + n, 128), acc);
    float bb = b1[co0 + n];
#pragma unroll
    for (int r = 0; r < 8; ++r) {
      int m = r + 8 * ((lane >> 4) & 1);
      float val = acc[r] + bb;
      val = 0.5f * val * (1.f + erff(val * 0.70710678f));   // exact GELU
      h1[m][co0 + n] = (_Float16)val;
    }
  }
  __syncthreads();

  // GEMM2: (16x128) x (128x64) + b2 + residual -> NCHW output
  int co0 = wave * 16;
  v8f acc = {};
#pragma unroll
  for (int c0 = 0; c0 < 128; c0 += 32)
    acc = wmma16(frag16(&h1[mrow][c0]), frag16s(w2 + c0 * 64 + co0 + n, 64), acc);
  float bb = b2[co0 + n];
#pragma unroll
  for (int r = 0; r < 8; ++r) {
    int m = r + 8 * ((lane >> 4) & 1);
    int t = t0 + m;
    int b = t / HWPIX, pix = t - b * HWPIX;
    int i2 = pix / SS, j2 = pix % SS;
    outp[(((size_t)b * 64 + co0 + n) * SS + i2) * SS + j2] = acc[r] + bb + sa[m][co0 + n];
  }
}

// ---------------- host-side launch ------------------------------------------
extern "C" void kernel_launch(void* const* d_in, const int* in_sizes, int n_in,
                              void* d_out, int out_size, void* d_ws, size_t ws_size,
                              hipStream_t stream) {
  (void)in_sizes; (void)n_in; (void)out_size; (void)ws_size;
  const float* xq      = (const float*)d_in[0];
  const float* xkv     = (const float*)d_in[1];
  const float* cw_s    = (const float*)d_in[2];
  const float* cb_s    = (const float*)d_in[3];
  const float* cw_l    = (const float*)d_in[4];
  const float* cb_l    = (const float*)d_in[5];
  const float* g_s     = (const float*)d_in[6];
  const float* b_s     = (const float*)d_in[7];
  const float* g_l     = (const float*)d_in[8];
  const float* b_l     = (const float*)d_in[9];
  const float* Wq      = (const float*)d_in[10];
  const float* bq      = (const float*)d_in[11];
  const float* Wk      = (const float*)d_in[12];
  const float* bk      = (const float*)d_in[13];
  const float* Wv      = (const float*)d_in[14];
  const float* bv      = (const float*)d_in[15];
  const float* Wo      = (const float*)d_in[16];
  const float* bo      = (const float*)d_in[17];
  const float* rpb     = (const float*)d_in[18];
  const float* g_n     = (const float*)d_in[19];
  const float* b_n     = (const float*)d_in[20];
  const float* W1      = (const float*)d_in[21];
  const float* b1      = (const float*)d_in[22];
  const float* W2      = (const float*)d_in[23];
  const float* b2      = (const float*)d_in[24];
  float* out = (float*)d_out;

  float* ws = (float*)d_ws;
  float* emb_q   = ws + 0 * (size_t)TOKC;
  float* emb_kv  = ws + 1 * (size_t)TOKC;
  float* qbuf    = ws + 2 * (size_t)TOKC;
  float* kbuf    = ws + 3 * (size_t)TOKC;
  float* vbuf    = ws + 4 * (size_t)TOKC;
  float* aobuf   = ws + 5 * (size_t)TOKC;
  float* attbuf  = ws + 6 * (size_t)TOKC;
  _Float16* hb   = (_Float16*)(ws + 7 * (size_t)TOKC);

  // 0) weights -> f16 (conv weights reordered for contiguous-K fragments)
  prep_weights<<<(NHALF + 255) / 256, 256, 0, stream>>>(
      cw_s, cw_l, Wq, Wk, Wv, Wo, W1, W2, hb);

  // 1) conv stems (implicit GEMM, WMMA)
  dim3 cgrid(BB * SS * 7);
  conv3x3_wmma<<<cgrid, 128, 0, stream>>>(xq,  hb + OFF_WS16, cb_s, emb_q);
  conv3x3_wmma<<<cgrid, 128, 0, stream>>>(xkv, hb + OFF_WL16, cb_l, emb_kv);

  // 2) LN + QKV projection (WMMA), q pre-scaled, head-major layout
  ln_qkv_wmma<<<NTOK / 16, 128, 0, stream>>>(
      emb_q, emb_kv, g_s, b_s, g_l, b_l,
      hb + OFF_WQ16, hb + OFF_WK16, hb + OFF_WV16,
      bq, bk, bv, qbuf, kbuf, vbuf);

  // 3) 7x7 neighborhood attention with online softmax
  na_attn<<<(BB * NHEADS * HWPIX) / 128, 128, 0, stream>>>(
      qbuf, kbuf, vbuf, rpb, aobuf);

  // 4) output projection + residual (WMMA)
  oproj_wmma<<<NTOK / 16, 128, 0, stream>>>(
      aobuf, hb + OFF_WO16, bo, emb_kv, attbuf);

  // 5) LN + MLP + residual, direct NCHW store (WMMA)
  mlp_wmma<<<NTOK / 16, 128, 0, stream>>>(
      attbuf, g_n, b_n, hb + OFF_W116, b1, hb + OFF_W216, b2, out);
}
